// CIFARMambaModel_77884936945961
// MI455X (gfx1250) — compile-verified, matrix operand order checked by reading, not run on hardware
//
#include <hip/hip_runtime.h>
#include <hip/hip_bf16.h>

typedef __attribute__((ext_vector_type(16))) _Float16 v16h;
typedef __attribute__((ext_vector_type(8)))  _Float16 v8h;
typedef __attribute__((ext_vector_type(8)))  float    v8f;
typedef __attribute__((ext_vector_type(4)))  int      v4i;

#define B_SZ   16
#define L_SZ   1024
#define IN_SZ  3
#define NLAY   2
#define DM     256
#define DOUT   10
#define DI     512          // D_INNER
#define DST    16           // D_STATE
#define DCONV  4
#define DTR    16           // DT_RANK
#define NDBL   48           // DTR + 2*DST
#define BT     (B_SZ * L_SZ)

#ifndef __has_builtin
#define __has_builtin(x) 0
#endif

#if __has_builtin(__builtin_amdgcn_global_load_async_to_lds_b128)
#define HAVE_ASYNC_LDS 1
#else
#define HAVE_ASYNC_LDS 0
#endif

#if HAVE_ASYNC_LDS
#if __has_builtin(__builtin_amdgcn_s_wait_asynccnt)
#define WAIT_ASYNC(n) __builtin_amdgcn_s_wait_asynccnt(n)
#else
#define WAIT_ASYNC(n) asm volatile("s_wait_asynccnt %0" :: "i"(n) : "memory")
#endif
#else
#define WAIT_ASYNC(n)
#endif

// Pointer types matching the builtin's signature: v4i in addrspace 1 / 3.
typedef __attribute__((address_space(1))) v4i* gv4i_t;
typedef __attribute__((address_space(3))) v4i* lv4i_t;

// Stage 16 bytes global -> LDS. Async (no VGPR roundtrip, ASYNCcnt-tracked)
// when the gfx1250 builtin is available, plain copy otherwise.
__device__ __forceinline__ void stage16(const _Float16* g, _Float16* l)
{
#if HAVE_ASYNC_LDS
    __builtin_amdgcn_global_load_async_to_lds_b128((gv4i_t)g, (lv4i_t)l, 0, 0);
#else
    *(v8h*)l = *(const v8h*)g;
#endif
}

// ---------------------------------------------------------------------------
// Weight convert (f32 -> f16) + transpose: in (K,N) -> out (N,K)
// ---------------------------------------------------------------------------
__global__ void wcvt_t_kernel(const float* __restrict__ w,
                              _Float16* __restrict__ wt, int K, int N)
{
    int idx = blockIdx.x * blockDim.x + threadIdx.x;
    if (idx >= K * N) return;
    int n = idx / K, k = idx % K;
    wt[idx] = (_Float16)w[(size_t)k * N + n];
}

// ---------------------------------------------------------------------------
// Input projection: h = x @ w_proj + b_proj   (K = 3, pure VALU)
// ---------------------------------------------------------------------------
__global__ void in_proj_kernel(const float* __restrict__ x,
                               const float* __restrict__ w,
                               const float* __restrict__ b,
                               float* __restrict__ h)
{
    int idx = blockIdx.x * blockDim.x + threadIdx.x;   // BT*DM
    int t = idx / DM, m = idx % DM;
    const float* xr = x + (size_t)t * IN_SZ;
    h[idx] = b[m] + xr[0] * w[0 * DM + m]
                  + xr[1] * w[1 * DM + m]
                  + xr[2] * w[2 * DM + m];
}

// ---------------------------------------------------------------------------
// RMSNorm (one wave32 per token, 8 elems/lane) -> f16 output
// ---------------------------------------------------------------------------
__global__ __launch_bounds__(256)
void rmsnorm_f16_kernel(const float* __restrict__ h,
                        const float* __restrict__ w,
                        _Float16* __restrict__ out)
{
    const int lane = threadIdx.x & 31;
    const int wv   = threadIdx.x >> 5;
    const int t    = blockIdx.x * 8 + wv;
    const float* row = h + (size_t)t * DM;
    float vals[8];
    float s = 0.f;
#pragma unroll
    for (int i = 0; i < 8; ++i) { float v = row[lane + i * 32]; vals[i] = v; s += v * v; }
#pragma unroll
    for (int off = 16; off > 0; off >>= 1) s += __shfl_xor(s, off);
    float rs = rsqrtf(s * (1.0f / DM) + 1e-6f);
    _Float16* orow = out + (size_t)t * DM;
#pragma unroll
    for (int i = 0; i < 8; ++i) { int m = lane + i * 32; orow[m] = (_Float16)(vals[i] * rs * w[m]); }
}

// ---------------------------------------------------------------------------
// WMMA GEMM: C(MxN,f32) = A(MxK,f16,row-major) * Bt(NxK,f16)^T
//
// Block = 8 waves covering a 64x128 C tile. The 64-row A tile is shared by
// all waves: staged cooperatively into LDS (double-buffered, async copy on
// gfx1250), killing the 8x redundant A reads. Row stride padded to 40 halves
// (80B) so the 64B-strided ds_load_b128 fragment reads are bank-conflict-free
// (20*r mod 64 is a permutation over r=0..15).
//
// Fragment layout per ISA 7.12.2: lane%16 = row(A)/col(B); half = lane>>4;
// elems 0..7 <- K = k0 + half*8 .. ; elems 8..15 <- K = k0 + 16 + half*8 ..
// ---------------------------------------------------------------------------
#define KCH   32
#define AROWS 64
#define APAD  40   // padded LDS row stride in halves

__global__ __launch_bounds__(256)
void wmma_gemm_f16f32(const _Float16* __restrict__ A,
                      const _Float16* __restrict__ Bt,
                      float* __restrict__ C,
                      int M, int N, int K, int accumulate)
{
    __shared__ __align__(16) _Float16 atile[2][AROWS * APAD];

    const int tid    = threadIdx.x;
    const int lane   = tid & 31;
    const int wave   = tid >> 5;
    const int lane16 = lane & 15;
    const int half   = lane >> 4;

    const int n0     = (blockIdx.y * 8 + wave) * 16;
    const bool active = (n0 < N);          // wave-uniform
    const int m0     = blockIdx.x * 64;

    // cooperative staging: 4 threads per row, 8 halves (16B) each
    const int srow = tid >> 2;
    const int scol = (tid & 3) * 8;
    const _Float16* gsrc = A + (size_t)(m0 + srow) * K + scol;
    _Float16* l0 = &atile[0][srow * APAD + scol];
    _Float16* l1 = &atile[1][srow * APAD + scol];

    const v8f vzero = {0.f, 0.f, 0.f, 0.f, 0.f, 0.f, 0.f, 0.f};
    v8f acc[4];
#pragma unroll
    for (int i = 0; i < 4; ++i) acc[i] = vzero;

    const int nclamp = active ? n0 : 0;
    const _Float16* brow = Bt + (size_t)(nclamp + lane16) * K + half * 8;

    const int nch = K / KCH;

    // prologue: stage chunk 0 into buffer 0
    stage16(gsrc, l0);

    for (int c = 0; c < nch; ++c) {
        const int buf = c & 1;
        const bool more = (c + 1 < nch);
        if (more) stage16(gsrc + (c + 1) * KCH, buf ? l0 : l1);

        if (more) { WAIT_ASYNC(1); } else { WAIT_ASYNC(0); }
        __syncthreads();                    // current buffer published

        if (active) {
            const int k0 = c * KCH;
            v8h blo = *(const v8h*)(brow + k0);
            v8h bhi = *(const v8h*)(brow + k0 + 16);
            v16h bfrag;
#pragma unroll
            for (int i = 0; i < 8; ++i) { bfrag[i] = blo[i]; bfrag[i + 8] = bhi[i]; }

            if (more) __builtin_prefetch(brow + k0 + KCH, 0, 0);

            const _Float16* at = atile[buf];
#pragma unroll
            for (int mt = 0; mt < 4; ++mt) {
                const _Float16* ar = at + (mt * 16 + lane16) * APAD + half * 8;
                v8h alo = *(const v8h*)(ar);
                v8h ahi = *(const v8h*)(ar + 16);
                v16h afrag;
#pragma unroll
                for (int i = 0; i < 8; ++i) { afrag[i] = alo[i]; afrag[i + 8] = ahi[i]; }
                acc[mt] = __builtin_amdgcn_wmma_f32_16x16x32_f16(
                    false, afrag, false, bfrag, (short)0, acc[mt], false, false);
            }
        }
        __syncthreads();                    // reads done before buf is overwritten
    }

    if (active) {
#pragma unroll
        for (int mt = 0; mt < 4; ++mt) {
#pragma unroll
            for (int r = 0; r < 8; ++r) {
                int row = m0 + mt * 16 + r + half * 8;   // lanes16-31 -> M+8
                int col = n0 + lane16;
                size_t idx = (size_t)row * N + col;
                float v = acc[mt][r];
                if (accumulate) C[idx] += v; else C[idx] = v;
            }
        }
    }
}

// ---------------------------------------------------------------------------
// Causal depthwise conv (width 4) + SiLU over xx half of xz -> f16
// ---------------------------------------------------------------------------
__global__ void conv_silu_kernel(const float* __restrict__ xz,
                                 const float* __restrict__ cw,
                                 const float* __restrict__ cb,
                                 _Float16* __restrict__ xx16)
{
    int idx = blockIdx.x * blockDim.x + threadIdx.x;   // B*L*DI
    int d  = idx % DI;
    int bt = idx / DI;
    int t  = bt % L_SZ;
    float a = cb[d];
#pragma unroll
    for (int j = 0; j < DCONV; ++j) {
        int tt = t - (DCONV - 1) + j;
        if (tt >= 0)
            a += cw[d * DCONV + j] * xz[(size_t)(bt - (t - tt)) * (2 * DI) + d];
    }
    a = a / (1.0f + __expf(-a));   // silu
    xx16[idx] = (_Float16)a;
}

// ---------------------------------------------------------------------------
// Selective scan: one block per batch element, one thread per inner channel.
// 16-wide SSM state lives in registers; (dt_in,B,C) row staged in LDS.
// Fuses dt = softplus(dbl[:, :16] @ w_dt + b_dt) and the SiLU(z) gate.
// ---------------------------------------------------------------------------
__global__ __launch_bounds__(512)
void scan_kernel(const float* __restrict__ dbl,      // BT x 48
                 const _Float16* __restrict__ xx16,  // BT x DI
                 const float* __restrict__ xz,       // BT x 2*DI (z at +DI)
                 const float* __restrict__ A_log,    // DI x DST
                 const float* __restrict__ Dp,       // DI
                 const float* __restrict__ w_dt,     // DTR x DI
                 const float* __restrict__ b_dt,     // DI
                 _Float16* __restrict__ y16)         // BT x DI
{
    __shared__ float srow[NDBL];
    const int d = threadIdx.x;
    const int b = blockIdx.x;

    float Aa[DST], hst[DST], wdt[DTR];
#pragma unroll
    for (int n = 0; n < DST; ++n) { Aa[n] = -__expf(A_log[d * DST + n]); hst[n] = 0.f; }
#pragma unroll
    for (int r = 0; r < DTR; ++r) wdt[r] = w_dt[r * DI + d];
    const float Dd = Dp[d], bd = b_dt[d];

    for (int t = 0; t < L_SZ; ++t) {
        const size_t bt = (size_t)b * L_SZ + t;
        __syncthreads();
        if (threadIdx.x < NDBL) srow[threadIdx.x] = dbl[bt * NDBL + threadIdx.x];
        __syncthreads();

        float dtin = bd;
#pragma unroll
        for (int r = 0; r < DTR; ++r) dtin += srow[r] * wdt[r];
        float dtv = (dtin > 20.f) ? dtin : __logf(1.f + __expf(dtin));  // softplus

        float u  = (float)xx16[bt * DI + d];
        float du = dtv * u;
        float y  = 0.f;
#pragma unroll
        for (int n = 0; n < DST; ++n) {
            float e = __expf(dtv * Aa[n]);
            hst[n] = e * hst[n] + du * srow[DTR + n];
            y += hst[n] * srow[DTR + DST + n];
        }
        y += u * Dd;
        float z = xz[bt * (2 * DI) + DI + d];
        y *= z / (1.f + __expf(-z));           // y * silu(z)
        y16[bt * DI + d] = (_Float16)y;
    }
}

// ---------------------------------------------------------------------------
// Final rmsnorm + mean-pool (atomic accumulate into pooled), then head
// ---------------------------------------------------------------------------
__global__ void zero_kernel(float* p, int n)
{
    int i = blockIdx.x * blockDim.x + threadIdx.x;
    if (i < n) p[i] = 0.f;
}

__global__ __launch_bounds__(256)
void final_pool_kernel(const float* __restrict__ h,
                       const float* __restrict__ w,
                       float* __restrict__ pooled)
{
    const int lane = threadIdx.x & 31;
    const int wv   = threadIdx.x >> 5;
    const int t    = blockIdx.x * 8 + wv;
    const int b    = t / L_SZ;
    const float* row = h + (size_t)t * DM;
    float vals[8];
    float s = 0.f;
#pragma unroll
    for (int i = 0; i < 8; ++i) { float v = row[lane + i * 32]; vals[i] = v; s += v * v; }
#pragma unroll
    for (int off = 16; off > 0; off >>= 1) s += __shfl_xor(s, off);
    float rs = rsqrtf(s * (1.0f / DM) + 1e-6f);
#pragma unroll
    for (int i = 0; i < 8; ++i) {
        int m = lane + i * 32;
        atomicAdd(&pooled[b * DM + m], vals[i] * rs * w[m] * (1.0f / L_SZ));
    }
}

__global__ void head_kernel(const float* __restrict__ pooled,
                            const float* __restrict__ wh,
                            const float* __restrict__ bh,
                            float* __restrict__ out)
{
    int tid = threadIdx.x;
    if (tid >= B_SZ * DOUT) return;
    int b = tid / DOUT, o = tid % DOUT;
    float s = bh[o];
    for (int m = 0; m < DM; ++m) s += pooled[b * DM + m] * wh[m * DOUT + o];
    out[tid] = s;
}

// ---------------------------------------------------------------------------
// Host orchestration
// ---------------------------------------------------------------------------
extern "C" void kernel_launch(void* const* d_in, const int* in_sizes, int n_in,
                              void* d_out, int out_size, void* d_ws, size_t ws_size,
                              hipStream_t stream)
{
    const float* x      = (const float*)d_in[0];
    const float* w_proj = (const float*)d_in[1];
    const float* b_proj = (const float*)d_in[2];
    const float* norm_w = (const float*)d_in[3];
    const float* w_in   = (const float*)d_in[4];
    const float* conv_w = (const float*)d_in[5];
    const float* conv_b = (const float*)d_in[6];
    const float* w_x    = (const float*)d_in[7];
    const float* w_dt   = (const float*)d_in[8];
    const float* b_dt   = (const float*)d_in[9];
    const float* A_log  = (const float*)d_in[10];
    const float* Dp     = (const float*)d_in[11];
    const float* w_out  = (const float*)d_in[12];
    const float* normf  = (const float*)d_in[13];
    const float* w_head = (const float*)d_in[14];
    const float* b_head = (const float*)d_in[15];

    char* ws = (char*)d_ws;
    size_t off = 0;
    auto carve = [&](size_t bytes) -> void* {
        void* p = ws + off;
        off = (off + bytes + 255) & ~(size_t)255;
        return p;
    };

    float*    h_res   = (float*)   carve((size_t)BT * DM * 4);
    _Float16* hn16    = (_Float16*)carve((size_t)BT * DM * 2);
    float*    xz      = (float*)   carve((size_t)BT * 2 * DI * 4);
    _Float16* xx16    = (_Float16*)carve((size_t)BT * DI * 2);
    float*    dbl     = (float*)   carve((size_t)BT * NDBL * 4);
    _Float16* y16     = (_Float16*)carve((size_t)BT * DI * 2);
    _Float16* w_in16  = (_Float16*)carve((size_t)NLAY * 2 * DI * DM * 2);  // (2DI, DM)
    _Float16* w_x16   = (_Float16*)carve((size_t)NLAY * NDBL * DI * 2);    // (48, DI)
    _Float16* w_out16 = (_Float16*)carve((size_t)NLAY * DM * DI * 2);      // (DM, DI)
    float*    pooled  = (float*)   carve((size_t)B_SZ * DM * 4);

    // --- weight prep: f32 -> f16 transposed ---
    for (int l = 0; l < NLAY; ++l) {
        wcvt_t_kernel<<<(DM * 2 * DI + 255) / 256, 256, 0, stream>>>(
            w_in + (size_t)l * DM * 2 * DI, w_in16 + (size_t)l * 2 * DI * DM, DM, 2 * DI);
        wcvt_t_kernel<<<(DI * NDBL + 255) / 256, 256, 0, stream>>>(
            w_x + (size_t)l * DI * NDBL, w_x16 + (size_t)l * NDBL * DI, DI, NDBL);
        wcvt_t_kernel<<<(DI * DM + 255) / 256, 256, 0, stream>>>(
            w_out + (size_t)l * DI * DM, w_out16 + (size_t)l * DM * DI, DI, DM);
    }

    // --- input projection ---
    in_proj_kernel<<<(BT * DM) / 256, 256, 0, stream>>>(x, w_proj, b_proj, h_res);

    // --- mamba layers ---
    for (int l = 0; l < NLAY; ++l) {
        rmsnorm_f16_kernel<<<BT / 8, 256, 0, stream>>>(h_res, norm_w + (size_t)l * DM, hn16);

        // xz = hn @ w_in      (M=BT, K=256, N=1024)
        wmma_gemm_f16f32<<<dim3(BT / 64, (2 * DI) / 128), 256, 0, stream>>>(
            hn16, w_in16 + (size_t)l * 2 * DI * DM, xz, BT, 2 * DI, DM, 0);

        conv_silu_kernel<<<(BT * DI) / 256, 256, 0, stream>>>(
            xz, conv_w + (size_t)l * DI * DCONV, conv_b + (size_t)l * DI, xx16);

        // dbl = xx @ w_x      (M=BT, K=512, N=48)
        wmma_gemm_f16f32<<<dim3(BT / 64, 1), 256, 0, stream>>>(
            xx16, w_x16 + (size_t)l * NDBL * DI, dbl, BT, NDBL, DI, 0);

        scan_kernel<<<B_SZ, DI, 0, stream>>>(
            dbl, xx16, xz,
            A_log + (size_t)l * DI * DST, Dp + (size_t)l * DI,
            w_dt + (size_t)l * DTR * DI, b_dt + (size_t)l * DI, y16);

        // h_res += y @ w_out  (M=BT, K=512, N=256)
        wmma_gemm_f16f32<<<dim3(BT / 64, DM / 128), 256, 0, stream>>>(
            y16, w_out16 + (size_t)l * DM * DI, h_res, BT, DM, DI, 1);
    }

    // --- final norm + mean pool + head ---
    zero_kernel<<<(B_SZ * DM + 255) / 256, 256, 0, stream>>>(pooled, B_SZ * DM);
    final_pool_kernel<<<BT / 8, 256, 0, stream>>>(h_res, normf, pooled);
    head_kernel<<<1, 256, 0, stream>>>(pooled, w_head, b_head, (float*)d_out);
}
